// TrellisMoELayer_7808250544397
// MI455X (gfx1250) — matrix operand order repaired and checked by reading, not compiled, not run on hardware
//
#include <hip/hip_runtime.h>
#include <hip/hip_bf16.h>

typedef _Float16 v16h __attribute__((ext_vector_type(16)));
typedef _Float16 v8h  __attribute__((ext_vector_type(8)));
typedef float    v8f  __attribute__((ext_vector_type(8)));
typedef float    v4f  __attribute__((ext_vector_type(4)));

#define MOE_H   2048
#define MOE_F   5632
#define MOE_E   8
#define MOE_FC  128               // F-chunk width (8 waves * 16 cols)
#define MOE_NCH (MOE_F / MOE_FC)  // 44
#define SA 2056                   // Atile row stride (halves), padded vs 2048
#define SH 136                    // h-chunk row stride (halves), padded vs 128
#define SO 2052                   // out row stride (floats), padded vs 2048

union H16 { v16h v; v8h h[2]; };

// Load 16 contiguous fp32 and convert to a 16-half fragment slice.
// B-fragment k order used here, same as A's (32x16): lane (l&15) = column N, lane>>4
// selects K block 0-15 / 16-31, K contiguous within the lane (2 per VGPR).
__device__ __forceinline__ v16h cvt_frag16(const float* __restrict__ p) {
    const v4f* q = (const v4f*)p;
    v4f f0 = q[0], f1 = q[1], f2 = q[2], f3 = q[3];
    v16h r;
#pragma unroll
    for (int i = 0; i < 4; ++i) {
        r[i]      = (_Float16)f0[i];
        r[4 + i]  = (_Float16)f1[i];
        r[8 + i]  = (_Float16)f2[i];
        r[12 + i] = (_Float16)f3[i];
    }
    return r;
}

// ---------------- Router: logits -> softmax -> top2 -> buckets ----------------
__global__ __launch_bounds__(256) void moe_router_kernel(
        const float* __restrict__ x, const float* __restrict__ rw,
        float* __restrict__ bw, int* __restrict__ bidx, int* __restrict__ cnt,
        int T) {
    extern __shared__ float rwS[];               // [E][H] = 64KB
    const int tid = threadIdx.x;
    for (int i = tid; i < MOE_E * MOE_H; i += 256) rwS[i] = rw[i];
    __syncthreads();

    const int wv = tid >> 5, lane = tid & 31;
    for (int t = blockIdx.x * 8 + wv; t < T; t += gridDim.x * 8) {
        float acc[MOE_E];
#pragma unroll
        for (int e = 0; e < MOE_E; ++e) acc[e] = 0.f;
        const float* xp = x + (size_t)t * MOE_H;
        for (int k = lane; k < MOE_H; k += 32) {
            float xv = xp[k];
#pragma unroll
            for (int e = 0; e < MOE_E; ++e) acc[e] += xv * rwS[e * MOE_H + k];
        }
#pragma unroll
        for (int off = 16; off > 0; off >>= 1) {
#pragma unroll
            for (int e = 0; e < MOE_E; ++e) acc[e] += __shfl_xor(acc[e], off, 32);
        }
        if (lane == 0) {
            float m = acc[0];
#pragma unroll
            for (int e = 1; e < MOE_E; ++e) m = fmaxf(m, acc[e]);
            float p[MOE_E];
#pragma unroll
            for (int e = 0; e < MOE_E; ++e) p[e] = __expf(acc[e] - m);
            // (softmax denominator cancels in the top-2 renormalization)
            int e1 = 0;
#pragma unroll
            for (int e = 1; e < MOE_E; ++e) if (p[e] > p[e1]) e1 = e;
            int e2 = (e1 == 0) ? 1 : 0;
#pragma unroll
            for (int e = 0; e < MOE_E; ++e)
                if (e != e1 && p[e] > p[e2]) e2 = e;
            float denom = p[e1] + p[e2];
            float w1 = p[e1] / denom, w2 = p[e2] / denom;
            int p1 = atomicAdd(&cnt[e1], 1);
            bidx[e1 * T + p1] = t;  bw[e1 * T + p1] = w1;
            int p2 = atomicAdd(&cnt[e2], 1);
            bidx[e2 * T + p2] = t;  bw[e2 * T + p2] = w2;
        }
    }
}

// ---------------- Fused expert: gate/up GEMM + SwiGLU + down GEMM -------------
__global__ __launch_bounds__(256) void moe_expert_kernel(
        const float* __restrict__ x,
        const float* __restrict__ Wg, const float* __restrict__ Wu,
        const float* __restrict__ Wd,
        const float* __restrict__ bw, const int* __restrict__ bidx,
        const int* __restrict__ cnt, float* __restrict__ out, int T) {
    extern __shared__ char smem[];
    _Float16* Atile = (_Float16*)smem;                        // 16 x SA halves (65792 B)
    float*    outS  = (float*)(smem + 65792);                 // 16 x SO floats (131328 B)
    _Float16* hS    = (_Float16*)(smem + 65792 + 131328);     // 16 x SH halves (4352 B)
    int*      tokS  = (int*)(smem + 65792 + 131328 + 4352);   // 16 ints
    float*    wS    = (float*)(smem + 65792 + 131328 + 4352 + 64); // 16 floats

    const int e = blockIdx.y;
    const int g = blockIdx.x;
    const int count = cnt[e];
    if (g * 16 >= count) return;

    const int tid = threadIdx.x;
    if (tid < 16) {
        int gp = g * 16 + tid;
        tokS[tid] = (gp < count) ? bidx[e * T + gp] : -1;
        wS[tid]   = (gp < count) ? bw[e * T + gp] : 0.f;
    }
    for (int i = tid; i < 16 * SO; i += 256) outS[i] = 0.f;
    __syncthreads();

    // Gather 16 token rows of x into LDS as f16 (A operand for gate/up)
    for (int idx = tid; idx < 16 * MOE_H; idx += 256) {
        int r = idx >> 11, c = idx & (MOE_H - 1);
        int t = tokS[r];
        float v = (t >= 0) ? x[(size_t)t * MOE_H + c] : 0.f;
        Atile[r * SA + c] = (_Float16)v;
    }
    __syncthreads();

    const float* WgE = Wg + (size_t)e * MOE_F * MOE_H;
    const float* WuE = Wu + (size_t)e * MOE_F * MOE_H;
    const float* WdE = Wd + (size_t)e * MOE_H * MOE_F;

    const int wv = tid >> 5, lane = tid & 31;
    const int col = lane & 15;       // A: row M / B: column N
    const int hi  = lane >> 4;       // lane-half selector

    for (int ch = 0; ch < MOE_NCH; ++ch) {
        // ---- phase 1: gate & up 16x16 tile per wave, K = H ----
        const int fbase = ch * MOE_FC + wv * 16;
        v8f ga = {}, ua = {};
        const _Float16* a_p  = Atile + col * SA + hi * 16;
        const float*    wg_p = WgE + (size_t)(fbase + col) * MOE_H + hi * 16;
        const float*    wu_p = WuE + (size_t)(fbase + col) * MOE_H + hi * 16;
#pragma unroll 8
        for (int ks = 0; ks < MOE_H / 32; ++ks) {
            H16 a;
            a.h[0] = *(const v8h*)(a_p);
            a.h[1] = *(const v8h*)(a_p + 8);
            a_p += 32;
            v16h bg = cvt_frag16(wg_p); wg_p += 32;
            v16h bu = cvt_frag16(wu_p); wu_p += 32;
            ga = __builtin_amdgcn_wmma_f32_16x16x32_f16(false, a.v, false, bg,
                                                        (short)0, ga, false, false);
            ua = __builtin_amdgcn_wmma_f32_16x16x32_f16(false, a.v, false, bu,
                                                        (short)0, ua, false, false);
        }
        // SwiGLU -> h chunk in LDS (C layout: m = v + hi*8, n = col)
#pragma unroll
        for (int v = 0; v < 8; ++v) {
            float gv = ga[v];
            float s  = gv / (1.f + __expf(-gv));
            float hv = s * ua[v];
            hS[(v + hi * 8) * SH + wv * 16 + col] = (_Float16)hv;
        }
        __syncthreads();

        // ---- phase 2: down-proj, out[16, wv*256 .. +256] += h_chunk @ Wd^T ----
        H16 af[4];
        const _Float16* hp = hS + col * SH + hi * 16;
#pragma unroll
        for (int ks = 0; ks < 4; ++ks) {
            af[ks].h[0] = *(const v8h*)(hp + ks * 32);
            af[ks].h[1] = *(const v8h*)(hp + ks * 32 + 8);
        }
#pragma unroll 2
        for (int j = 0; j < 16; ++j) {
            const int nb = wv * 256 + j * 16;
            float* op = outS + (size_t)(hi * 8) * SO + nb + col;
            v8f c;
#pragma unroll
            for (int v = 0; v < 8; ++v) c[v] = op[v * SO];
            const float* wd_p = WdE + (size_t)(nb + col) * MOE_F + ch * MOE_FC + hi * 16;
#pragma unroll
            for (int ks = 0; ks < 4; ++ks) {
                v16h b = cvt_frag16(wd_p + ks * 32);
                c = __builtin_amdgcn_wmma_f32_16x16x32_f16(false, af[ks].v, false, b,
                                                           (short)0, c, false, false);
            }
#pragma unroll
            for (int v = 0; v < 8; ++v) op[v * SO] = c[v];
        }
        __syncthreads();
    }

    // ---- combine-weighted accumulate into global output ----
    for (int idx = tid; idx < 16 * MOE_H; idx += 256) {
        int r = idx >> 11, c = idx & (MOE_H - 1);
        int t = tokS[r];
        if (t >= 0)
            unsafeAtomicAdd(out + (size_t)t * MOE_H + c, outS[r * SO + c] * wS[r]);
    }
}

extern "C" void kernel_launch(void* const* d_in, const int* in_sizes, int n_in,
                              void* d_out, int out_size, void* d_ws, size_t ws_size,
                              hipStream_t stream) {
    const float* x  = (const float*)d_in[0];
    const float* rw = (const float*)d_in[1];
    const float* Wg = (const float*)d_in[2];
    const float* Wu = (const float*)d_in[3];
    const float* Wd = (const float*)d_in[4];
    float* out = (float*)d_out;

    const int T = in_sizes[0] / MOE_H;           // 8192 tokens

    // workspace: bucket weights [E*T] f32, bucket token ids [E*T] i32, counts [E]
    float* bw  = (float*)d_ws;
    int* bidx  = (int*)((char*)d_ws + (size_t)MOE_E * T * 4);
    int* cnt   = (int*)((char*)d_ws + (size_t)MOE_E * T * 8);

    hipMemsetAsync(cnt, 0, MOE_E * sizeof(int), stream);
    hipMemsetAsync(d_out, 0, (size_t)out_size * sizeof(float), stream);

    moe_router_kernel<<<64, 256, MOE_E * MOE_H * sizeof(float), stream>>>(
        x, rw, bw, bidx, cnt, T);

    const size_t smem = 65792 + 131328 + 4352 + 64 + 64;   // 201600 B < 320KB/WGP
    dim3 grid((T + 15) / 16, MOE_E);
    moe_expert_kernel<<<grid, 256, smem, stream>>>(
        x, Wg, Wu, Wd, bw, bidx, cnt, out, T);
}
